// HypergraphConv_61194694033668
// MI455X (gfx1250) — matrix operand order repaired and checked by reading, not compiled, not run on hardware
//
#include <hip/hip_runtime.h>

typedef __attribute__((ext_vector_type(16))) __bf16 v16bf;
typedef __attribute__((ext_vector_type(8)))  __bf16 v8bf;
typedef __attribute__((ext_vector_type(8)))  float  v8f;
typedef __attribute__((ext_vector_type(4)))  float  f32x4;

#define NN     20000   // nodes
#define EE     8000    // hyperedges
#define FDIM   128     // features
#define MTILE  64
#define KTILE  32
#define APITCH 40      // halves: 80B rows, 16B aligned, bank-spread
#define BPITCH 40
#define YPITCH 136     // halves: 272B rows, 16B aligned, conflict-free col reads
#define WPITCH 128
#define EPSV   1e-6f

// ---------------- gfx1250 async global->LDS copy (ASYNCcnt path) ----------
__device__ __forceinline__ void async_copy_b128(unsigned lds_byte_addr,
                                                const void* gptr) {
  asm volatile("global_load_async_to_lds_b128 %0, %1, off"
               :: "v"(lds_byte_addr), "v"((unsigned long long)(size_t)gptr)
               : "memory");
}
__device__ __forceinline__ void wait_async0() {
  asm volatile("s_wait_asynccnt 0x0" ::: "memory");
}

// ---------------- WMMA fragment loaders (CDNA5 ISA 7.12.2, wave32) --------
__device__ __forceinline__ v16bf load_a_frag(const __bf16* rowbase, int hi) {
  union { v16bf v; v8bf h[2]; } u;
  u.h[0] = *(const v8bf*)(rowbase + hi * 8);
  u.h[1] = *(const v8bf*)(rowbase + 16 + hi * 8);
  return u.v;
}
__device__ __forceinline__ v16bf load_b_frag(const __bf16* rowbase, int hi) {
  union { v16bf v; v8bf h[2]; } u;
  u.h[0] = *(const v8bf*)(rowbase + hi * 8);
  u.h[1] = *(const v8bf*)(rowbase + 16 + hi * 8);
  return u.v;
}
__device__ __forceinline__ v8f wmma_bf16(v16bf a, v16bf b, v8f c) {
  return __builtin_amdgcn_wmma_f32_16x16x32_bf16(false, a, false, b, (short)0, c,
                                                 false, false);
}

// =====================================================================
// Kernel 1: MT[f][e] (bf16) = inv_de[e] * (H^T @ X)[e][f]   (transposed out)
// =====================================================================
__global__ __launch_bounds__(256) void hg_gemm1(const float* __restrict__ H,
                                                const float* __restrict__ X,
                                                __bf16* __restrict__ MT) {
  __shared__ __align__(16) unsigned char smem[17408 + 256 * 4 + MTILE * 4];
  __bf16* Alds = (__bf16*)smem;                         // [64][40]
  __bf16* Blds = (__bf16*)(smem + MTILE * APITCH * 2);  // [128][40]
  __bf16* Llds = (__bf16*)smem;                         // epilogue overlay [64][136]
  float*  red  = (float*)(smem + 17408);
  float*  inv  = red + 256;

  const int t    = threadIdx.x;
  const int e0   = blockIdx.x * MTILE;
  const int lane = t & 31, w = t >> 5;
  const int wm = w & 3, wN = w >> 2;
  const int l = lane & 15, hi = lane >> 4;

  const int sa_e = t & 63, sa_g = t >> 6;      // A: 4 k-groups x 8
  const int sb_f = t & 127, sb_kh = t >> 7;    // B: 2 k-groups x 16

  const float* pA = H + (size_t)(sa_g * 8) * EE + e0 + sa_e;
  const float* pB = X + (size_t)(sb_kh * 16) * FDIM + sb_f;
  __bf16* ALp = &Alds[sa_e * APITCH + sa_g * 8];
  __bf16* BLp = &Blds[sb_f * BPITCH + sb_kh * 16];

  v8f acc[4] = {{0,0,0,0,0,0,0,0},{0,0,0,0,0,0,0,0},
                {0,0,0,0,0,0,0,0},{0,0,0,0,0,0,0,0}};
  float csum = 0.f;

  constexpr int ITERS = NN / KTILE;            // 625 (exact)
  for (int it = 0; it < ITERS; ++it) {
    __syncthreads();
    // A = H^T tile: 8 column loads, immediate offsets, NT (stream H once)
    v8bf av;
#pragma unroll
    for (int j = 0; j < 8; ++j) {
      float v = __builtin_nontemporal_load(pA + (size_t)j * EE);
      csum += v;
      av[j] = (__bf16)v;
    }
    *(v8bf*)ALp = av;
    if (it + 1 < ITERS) __builtin_prefetch(pA + (size_t)KTILE * EE, 0, 0);
    // B = X tile transposed to [f][k] (X stays L2-resident: temporal loads)
    v8bf b0, b1;
#pragma unroll
    for (int j = 0; j < 8; ++j) b0[j] = (__bf16)pB[(size_t)j * FDIM];
#pragma unroll
    for (int j = 0; j < 8; ++j) b1[j] = (__bf16)pB[(size_t)(j + 8) * FDIM];
    ((v8bf*)BLp)[0] = b0;
    ((v8bf*)BLp)[1] = b1;
    pA += (size_t)KTILE * EE;
    pB += KTILE * FDIM;
    __syncthreads();

    v16bf a = load_a_frag(&Alds[(wm * 16 + l) * APITCH], hi);
#pragma unroll
    for (int j = 0; j < 4; ++j) {
      v16bf b = load_b_frag(&Blds[(wN * 64 + j * 16 + l) * BPITCH], hi);
      acc[j] = wmma_bf16(a, b, acc[j]);
    }
  }

  // colsum -> inv_de
  __syncthreads();
  red[t] = csum;
  __syncthreads();
  if (t < 64)
    inv[t] = 1.f / (red[t] + red[t + 64] + red[t + 128] + red[t + 192] + EPSV);
  __syncthreads();

  // scale rows, park in LDS for transposed coalesced writeout
#pragma unroll
  for (int j = 0; j < 4; ++j) {
    int col = wN * 64 + j * 16 + l;
#pragma unroll
    for (int v = 0; v < 8; ++v) {
      int row = wm * 16 + hi * 8 + v;
      Llds[row * YPITCH + col] = (__bf16)(acc[j][v] * inv[row]);
    }
  }
  __syncthreads();

  // MT[f][e0+..]: 16B stores, 64B contiguous per thread
  const int f = t >> 1, h = t & 1;
  __bf16* dst = MT + (size_t)f * EE + e0 + h * 32;
#pragma unroll
  for (int g = 0; g < 4; ++g) {
    v8bf o;
#pragma unroll
    for (int s = 0; s < 8; ++s) o[s] = Llds[(h * 32 + g * 8 + s) * YPITCH + f];
    *(v8bf*)(dst + g * 8) = o;
  }
}

// =====================================================================
// Kernel 2: out[n][128] (f32) = (inv_dv[n] * (H @ M)) @ W,  all fused
// =====================================================================
__global__ __launch_bounds__(256) void hg_gemm2(const float* __restrict__ H,
                                                const __bf16* __restrict__ MT,
                                                const float* __restrict__ W,
                                                float* __restrict__ out) {
  __shared__ __align__(16) unsigned char smem[17408 + FDIM * WPITCH * 2 +
                                              256 * 4 + MTILE * 4];
  __bf16* Alds = (__bf16*)smem;                         // [64][40]
  __bf16* Blds = (__bf16*)(smem + MTILE * APITCH * 2);  // [128][40]
  __bf16* Ylds = (__bf16*)smem;                         // phase-2 overlay [64][136]
  __bf16* Wlds = (__bf16*)(smem + 17408);               // [fout][fin], pitch 128
  float*  red  = (float*)(smem + 17408 + FDIM * WPITCH * 2);
  float*  inv  = red + 256;

  const int t    = threadIdx.x;
  const int i0   = blockIdx.x * MTILE;
  const int lane = t & 31, w = t >> 5;
  const int wm = w & 3, wN = w >> 2;
  const int l = lane & 15, hi = lane >> 4;

  // stage weight once: W(fin,fout) row-major -> Wlds[fout][fin] bf16
#pragma unroll 4
  for (int it = 0; it < 64; ++it) {
    int idx = t + it * 256;
    Wlds[(idx & 127) * WPITCH + (idx >> 7)] = (__bf16)W[idx];
  }

  // A staging: row = t>>2, 4 k-groups x 8 consecutive f32
  const int sa_i = t >> 2, sa_kg = t & 3;
  const bool rok = (i0 + sa_i) < NN;
  const size_t grow = rok ? (size_t)(i0 + sa_i) : (size_t)(NN - 1);  // clamp
  const float* pH = H + grow * EE + sa_kg * 8;
  __bf16* ALp = &Alds[sa_i * APITCH + sa_kg * 8];
  // B staging: async b128 from MT[f][k] (bf16, contiguous K runs)
  const __bf16* pM = MT + (size_t)(t >> 1) * EE + (t & 1) * 16;
  const unsigned ldsB =
      (unsigned)(size_t)&Blds[(t >> 1) * BPITCH + (t & 1) * 16];

  v8f acc[4] = {{0,0,0,0,0,0,0,0},{0,0,0,0,0,0,0,0},
                {0,0,0,0,0,0,0,0},{0,0,0,0,0,0,0,0}};
  float rsum = 0.f;
  const f32x4 zf4 = {0.f, 0.f, 0.f, 0.f};

  constexpr int ITERS = EE / KTILE;            // 250 (exact)
  for (int it = 0; it < ITERS; ++it) {
    __syncthreads();
    // kick off async B copy first, overlap with A conversion work
    async_copy_b128(ldsB, pM);
    async_copy_b128(ldsB + 16, pM + 8);
    pM += KTILE;
    // A = H rows: two 16B NT loads, branchless tail masking
    f32x4 v0 = __builtin_nontemporal_load((const f32x4*)pH);
    f32x4 v1 = __builtin_nontemporal_load((const f32x4*)pH + 1);
    v0 = rok ? v0 : zf4;
    v1 = rok ? v1 : zf4;
    rsum += v0.x + v0.y + v0.z + v0.w + v1.x + v1.y + v1.z + v1.w;
    v8bf av;
    av[0] = (__bf16)v0.x; av[1] = (__bf16)v0.y;
    av[2] = (__bf16)v0.z; av[3] = (__bf16)v0.w;
    av[4] = (__bf16)v1.x; av[5] = (__bf16)v1.y;
    av[6] = (__bf16)v1.z; av[7] = (__bf16)v1.w;
    *(v8bf*)ALp = av;
    if (it + 1 < ITERS) __builtin_prefetch(pH + KTILE, 0, 0);
    pH += KTILE;
    wait_async0();
    __syncthreads();

    v16bf a = load_a_frag(&Alds[(wm * 16 + l) * APITCH], hi);
#pragma unroll
    for (int j = 0; j < 4; ++j) {
      v16bf b = load_b_frag(&Blds[(wN * 64 + j * 16 + l) * BPITCH], hi);
      acc[j] = wmma_bf16(a, b, acc[j]);
    }
  }

  // rowsum -> inv_dv   (red index == row*4 + kgroup == t)
  __syncthreads();
  red[t] = rsum;
  __syncthreads();
  if (t < 64)
    inv[t] = 1.f / (red[t * 4] + red[t * 4 + 1] + red[t * 4 + 2] +
                    red[t * 4 + 3] + EPSV);
  __syncthreads();

  // scale Y, park bf16 in LDS for the weight GEMM
#pragma unroll
  for (int j = 0; j < 4; ++j) {
    int col = wN * 64 + j * 16 + l;
#pragma unroll
    for (int v = 0; v < 8; ++v) {
      int row = wm * 16 + hi * 8 + v;
      Ylds[row * YPITCH + col] = (__bf16)(acc[j][v] * inv[row]);
    }
  }
  __syncthreads();

  // phase 2: out_tile = Y(64x128) @ W(128x128), K=128 in 4 chunks
  v8f acc2[4] = {{0,0,0,0,0,0,0,0},{0,0,0,0,0,0,0,0},
                 {0,0,0,0,0,0,0,0},{0,0,0,0,0,0,0,0}};
#pragma unroll
  for (int kk = 0; kk < 4; ++kk) {
    v16bf a2 = load_a_frag(&Ylds[(wm * 16 + l) * YPITCH + kk * 32], hi);
#pragma unroll
    for (int j = 0; j < 4; ++j) {
      v16bf b2 = load_b_frag(&Wlds[(wN * 64 + j * 16 + l) * WPITCH + kk * 32], hi);
      acc2[j] = wmma_bf16(a2, b2, acc2[j]);
    }
  }

  // guarded f32 writeback (tail rows only diverge in the last block)
#pragma unroll
  for (int j = 0; j < 4; ++j) {
    int col = wN * 64 + j * 16 + l;
#pragma unroll
    for (int v = 0; v < 8; ++v) {
      int row = wm * 16 + hi * 8 + v;
      if (i0 + row < NN) out[(size_t)(i0 + row) * FDIM + col] = acc2[j][v];
    }
  }
}

extern "C" void kernel_launch(void* const* d_in, const int* in_sizes, int n_in,
                              void* d_out, int out_size, void* d_ws, size_t ws_size,
                              hipStream_t stream) {
  const float* X = (const float*)d_in[0];   // (NN,128)
  const float* H = (const float*)d_in[1];   // (NN,EE)
  const float* W = (const float*)d_in[2];   // (128,128)
  float* out = (float*)d_out;               // (NN,128) f32
  __bf16* MT = (__bf16*)d_ws;               // (128,EE) bf16, 2 MB

  hg_gemm1<<<dim3(EE / MTILE), dim3(256), 0, stream>>>(H, X, MT);
  hg_gemm2<<<dim3((NN + MTILE - 1) / MTILE), dim3(256), 0, stream>>>(H, MT, W, out);
}
